// SelfAttention_32667521253551
// MI455X (gfx1250) — matrix operand order repaired and checked by reading, not compiled, not run on hardware
//
#include <hip/hip_runtime.h>
#include <hip/hip_bf16.h>

// ---------------------------------------------------------------------------
// Fused "distance-decay pseudo-softmax attention" for B=32, S=1024, D=1024.
// All three GEMMs run on v_wmma_f32_16x16x32_bf16 (fp32 accumulate).
// ---------------------------------------------------------------------------

typedef __bf16 bf16;
typedef __attribute__((ext_vector_type(16))) __bf16 v16bf;
typedef __attribute__((ext_vector_type(8)))  __bf16 v8bf;
typedef __attribute__((ext_vector_type(8)))  float  v8f;

#define S_DIM 1024
#define D_DIM 1024
#define EPSF  1e-5f

// A-matrix fragment (16x32 bf16): lane holds row m = lane&15.
// Per ISA 7.12.2: VGPR v<4 -> K = 8h + 2v,2v+1 ; v>=4 -> K = 16 + 8h + ...
// => lane needs bf16 at k0+8h+(0..7) and k0+16+8h+(0..7) (two 16B chunks).
__device__ __forceinline__ v16bf load_a_frag(const bf16* __restrict__ rowp, int k0, int h) {
  v8bf lo = *reinterpret_cast<const v8bf*>(rowp + k0 + 8 * h);
  v8bf hi = *reinterpret_cast<const v8bf*>(rowp + k0 + 16 + 8 * h);
  return __builtin_shufflevector(lo, hi, 0,1,2,3,4,5,6,7,8,9,10,11,12,13,14,15);
}

// B-matrix fragment (32x16 bf16): lane holds column n = lane&15,
// K = 8h + (0..7) and 16 + 8h + (0..7) (two 16B chunks, same as A). rowp points at B^T row n.
__device__ __forceinline__ v16bf load_b_frag(const bf16* __restrict__ rowp, int k0, int h) {
  v8bf lo = *reinterpret_cast<const v8bf*>(rowp + k0 + 8 * h);
  v8bf hi = *reinterpret_cast<const v8bf*>(rowp + k0 + 16 + 8 * h);
  return __builtin_shufflevector(lo, hi, 0,1,2,3,4,5,6,7,8,9,10,11,12,13,14,15);
}

// --------------------------- prep kernels ----------------------------------

__global__ void cvt_bf16_kernel(const float* __restrict__ in, bf16* __restrict__ out, int n) {
  int i = blockIdx.x * blockDim.x + threadIdx.x;
  if (i < n) out[i] = (bf16)in[i];
}

// xbT[b][d][t] = (bf16) x[b][t][d]   (coalesced read, strided write; one-shot)
__global__ void transpose_bf16_kernel(const float* __restrict__ x, bf16* __restrict__ xbT) {
  size_t i = (size_t)blockIdx.x * blockDim.x + threadIdx.x;   // b*S*D + t*D + d
  int d = (int)(i & (D_DIM - 1));
  int t = (int)((i >> 10) & (S_DIM - 1));
  int b = (int)(i >> 20);
  xbT[((size_t)b * D_DIM + d) * S_DIM + t] = (bf16)x[i];
}

// ow[b,s] = gp*(g1+g2) + (1-gp)*(p3+p4)
__global__ void opinion_kernel(const float* __restrict__ g, const float* __restrict__ p,
                               const float* __restrict__ gp, float* __restrict__ ow, int n) {
  int i = blockIdx.x * blockDim.x + threadIdx.x;
  if (i < n) {
    float q = gp[0];
    ow[i] = q * (g[i * 5 + 1] + g[i * 5 + 2]) + (1.0f - q) * (p[i * 5 + 3] + p[i * 5 + 4]);
  }
}

// --------------------------- fused attention -------------------------------
// One block = 16 query rows of one batch. 8 waves (256 threads, wave32).
// Wave w owns e-range / d-range [128w, 128w+128) and key col-tile w within
// each 128-key iteration.

__global__ __launch_bounds__(256) void attn_fused_kernel(
    const bf16* __restrict__ xb,   // [B,S,D] bf16 row-major
    const bf16* __restrict__ xbT,  // [B,D,S] bf16 (x transposed per batch)
    const bf16* __restrict__ Wb,   // [D,D]   bf16 row-major (W[e][d])
    const float* __restrict__ bias,
    const float* __restrict__ ow,  // [B,S] opinion weights
    float* __restrict__ out)       // [B,S,D] fp32
{
  __shared__ alignas(16) bf16 sXt[16 * 1032];  // x_tran tile, bf16, padded rows
  __shared__ alignas(16) bf16 sP[16 * 136];    // P tile (16 q x 128 keys), padded
  __shared__ float sRow[16];                   // row sums

  const int tid  = threadIdx.x;
  const int w    = tid >> 5;
  const int lane = tid & 31;
  const int h    = lane >> 4;   // lane half
  const int ln   = lane & 15;

  const int b  = blockIdx.x >> 6;          // 64 query tiles per batch
  const int s0 = (blockIdx.x & 63) << 4;

  const size_t rowbase = (size_t)b * S_DIM + s0;
  const bf16* xrow = xb + (rowbase + ln) * D_DIM;   // A rows (query rows)

  if (tid < 16) sRow[tid] = 0.0f;

  // ---- Step A: x_tran tile = x_rows @ W^T + b  -> sXt (bf16) ----
  for (int nt = 0; nt < 8; ++nt) {
    const int e0 = w * 128 + nt * 16;
    const bf16* wrow = Wb + (size_t)(e0 + ln) * D_DIM;
    v8f acc = {};
    for (int k0 = 0; k0 < D_DIM; k0 += 32) {
      v16bf a  = load_a_frag(xrow, k0, h);
      v16bf bm = load_b_frag(wrow, k0, h);
      acc = __builtin_amdgcn_wmma_f32_16x16x32_bf16(false, a, false, bm,
                                                    (short)0, acc, false, false);
    }
    const float bv = bias[e0 + ln];
#pragma unroll
    for (int r = 0; r < 8; ++r) {
      const int m = r + 8 * h;               // C layout: M = vgpr + 8*half
      sXt[m * 1032 + e0 + ln] = (bf16)(acc[r] + bv);
    }
  }
  __syncthreads();

  float rs[8];
  float owv[8];
#pragma unroll
  for (int r = 0; r < 8; ++r) {
    rs[r]  = 0.0f;
    owv[r] = ow[rowbase + r + 8 * h];
  }

  v8f oacc[8];
#pragma unroll
  for (int nt = 0; nt < 8; ++nt) oacc[nt] = (v8f){};

  const bf16* xtrow = sXt + ln * 1032;   // A rows of x_tran tile
  const bf16* prow  = sP  + ln * 136;    // A rows of P tile

  // ---- Stream over keys, 128 at a time ----
  for (int t0 = 0; t0 < S_DIM; t0 += 128) {
    // GEMM1: this wave's 16-key column tile of S = x_tran @ x^T
    const int tcol = t0 + w * 16;
    const bf16* krow = xb + ((size_t)b * S_DIM + tcol + ln) * D_DIM;
    v8f sacc = {};
    for (int k0 = 0; k0 < D_DIM; k0 += 32) {
      v16bf a  = load_a_frag(xtrow, k0, h);
      v16bf bm = load_b_frag(krow, k0, h);
      sacc = __builtin_amdgcn_wmma_f32_16x16x32_bf16(false, a, false, bm,
                                                     (short)0, sacc, false, false);
    }
    __syncthreads();   // previous iteration's GEMM2 finished reading sP

    // elementwise: distance decay, opinion weight, exp(tanh), diag mask
#pragma unroll
    for (int r = 0; r < 8; ++r) {
      const int m = r + 8 * h;
      const int i = s0 + m;
      const int j = tcol + ln;
      float wv  = sacc[r];
      float loc = fabsf((float)(i - j));
      wv *= 1.0f / (loc + EPSF);
      wv *= owv[r];
      wv  = __expf(tanhf(wv));
      if (i == j) wv = 0.0f;
      rs[r] += wv;
      sP[m * 136 + w * 16 + ln] = (bf16)wv;
    }
    __syncthreads();   // sP ready for all waves

    // GEMM2: oacc += P @ x  over this 128-key block, wave's d-range
#pragma unroll
    for (int nt = 0; nt < 8; ++nt) {
      const int d0 = w * 128 + nt * 16;
      const bf16* trow = xbT + ((size_t)b * D_DIM + d0 + ln) * S_DIM + t0;
#pragma unroll
      for (int kk = 0; kk < 4; ++kk) {
        v16bf a  = load_a_frag(prow, kk * 32, h);
        v16bf bm = load_b_frag(trow, kk * 32, h);
        oacc[nt] = __builtin_amdgcn_wmma_f32_16x16x32_bf16(false, a, false, bm,
                                                           (short)0, oacc[nt], false, false);
      }
    }
  }

  // ---- row-sum reduction and normalized store ----
#pragma unroll
  for (int r = 0; r < 8; ++r) atomicAdd(&sRow[r + 8 * h], rs[r]);
  __syncthreads();

#pragma unroll
  for (int nt = 0; nt < 8; ++nt) {
    const int d = w * 128 + nt * 16 + ln;
#pragma unroll
    for (int r = 0; r < 8; ++r) {
      const int m = r + 8 * h;
      out[(rowbase + m) * D_DIM + d] = oacc[nt][r] / (sRow[m] + EPSF);
    }
  }
}

// --------------------------- host launcher ---------------------------------

extern "C" void kernel_launch(void* const* d_in, const int* in_sizes, int n_in,
                              void* d_out, int out_size, void* d_ws, size_t ws_size,
                              hipStream_t stream) {
  const float* x    = (const float*)d_in[0];  // [32,1024,1024]
  const float* W    = (const float*)d_in[1];  // [1024,1024]
  const float* bias = (const float*)d_in[2];  // [1024]
  const float* g    = (const float*)d_in[3];  // [32,1024,5]
  const float* p    = (const float*)d_in[4];  // [32,1024,5]
  const float* gp   = (const float*)d_in[5];  // scalar
  float* out        = (float*)d_out;

  char* ws   = (char*)d_ws;
  bf16* xb   = (bf16*)(ws);                                   // 64 MB
  bf16* xbT  = (bf16*)(ws + (size_t)64 * 1024 * 1024);        // 64 MB
  bf16* Wb   = (bf16*)(ws + (size_t)128 * 1024 * 1024);       // 2 MB
  float* ow  = (float*)(ws + (size_t)132 * 1024 * 1024);      // 128 KB

  const int NX = 32 * 1024 * 1024;   // x elements
  const int NW = 1024 * 1024;        // W elements
  const int NO = 32 * 1024;          // opinion rows

  cvt_bf16_kernel<<<NX / 256, 256, 0, stream>>>(x, xb, NX);
  transpose_bf16_kernel<<<NX / 256, 256, 0, stream>>>(x, xbT);
  cvt_bf16_kernel<<<NW / 256, 256, 0, stream>>>(W, Wb, NW);
  opinion_kernel<<<NO / 256, 256, 0, stream>>>(g, p, gp, ow, NO);

  attn_fused_kernel<<<2048, 256, 0, stream>>>(xb, xbT, Wb, bias, ow, out);
}
